// RRNNGRUCell_30331059045151
// MI455X (gfx1250) — compile-verified, matrix operand order checked by reading, not run on hardware
//
#include <hip/hip_runtime.h>

typedef __attribute__((ext_vector_type(16))) __bf16        bf16x16;
typedef __attribute__((ext_vector_type(8)))  float         floatx8;
typedef __attribute__((ext_vector_type(4)))  unsigned int  u32x4;

#define HID   256
#define BATCH 16384
#define TM    32          // rows per block
#define PITCH 264         // LDS row pitch (bf16 elems), conflict-spreading
#define NTILE 16          // 256 / 16 N-tiles

// workspace byte offsets
#define WPACK_OFF 0
#define WPACK_BYTES (8*16*8*32*32)        // 8 mats * 16 tiles * 8 ksteps * 32 lanes * 32B = 1 MiB
#define LWS_OFF   (WPACK_OFF + WPACK_BYTES)
#define RWS_OFF   (LWS_OFF + 4096)
#define BWS_OFF   (RWS_OFF + 4096)
#define SUMWS_OFF (BWS_OFF + 16)
#define GSC_OFF   (SUMWS_OFF + 16)

// output element offsets (f32 elements)
#define OUT_GS    (BATCH*HID)             // 4194304
#define OUT_GNODE (OUT_GS + 9)
#define OUT_MARG  (OUT_GNODE + (long long)BATCH*9*HID)

#define SMEM_BYTES 115712

// explicit global-address-space access helpers (guarantee global_load/store,
// tracked by LOADcnt/STOREcnt only — no flat_* / DScnt coupling)
__device__ __forceinline__ u32x4 ldg_b128(const void* p) {
  return *(__attribute__((address_space(1))) const u32x4*)p;
}
__device__ __forceinline__ void stg_b32(float* p, float v) {
  *(__attribute__((address_space(1))) float*)p = v;
}

// ---------------------------------------------------------------- pack weights
// B-fragment layout (16-bit 32x16): lane%16 = N col, lane/16 = half,
// vgpr v pair base k = (v<4?0:16) + half*8 + (v&3)*2  (low 16 bits = even k)
__global__ void pack_weights(const float* __restrict__ L, const float* __restrict__ R,
                             uint32_t* __restrict__ wpack) {
  int id = blockIdx.x * blockDim.x + threadIdx.x;   // 32768 threads
  int lane = id & 31, s = (id >> 5) & 7, t = (id >> 8) & 15, wi = id >> 12;
  const float* W = (wi < 4) ? (L + wi * HID * HID) : (R + (wi - 4) * HID * HID);
  int half = lane >> 4;
  int n = t * 16 + (lane & 15);
  u32x4 lo, hi;
#pragma unroll
  for (int v = 0; v < 8; ++v) {
    int kb = s * 32 + ((v < 4) ? 0 : 16) + half * 8 + (v & 3) * 2;
    union { uint32_t u; __bf16 hh[2]; } c;
    c.hh[0] = (__bf16)W[(kb + 0) * HID + n];
    c.hh[1] = (__bf16)W[(kb + 1) * HID + n];
    if (v < 4) lo[v] = c.u; else hi[v - 4] = c.u;
  }
  uint32_t* dst = wpack + ((((wi * 16 + t) * 8 + s) * 32) + lane) * 8;
  *(u32x4*)dst       = lo;
  *(u32x4*)(dst + 4) = hi;
}

// ---------------------------------------------------------------- prep vectors
__global__ void prep_vecs(const float* __restrict__ L, const float* __restrict__ R,
                          const float* __restrict__ b, const float* __restrict__ Ws,
                          float* __restrict__ lws, float* __restrict__ rws,
                          float* __restrict__ bws, float* __restrict__ sumws,
                          float* __restrict__ gsc) {
  int gid = blockIdx.x * 256 + threadIdx.x;
  if (gid < 2048) {
    int which = gid >> 10;                  // 0 -> L, 1 -> R
    int k = (gid >> 8) & 3, hh = gid & 255;
    const float* W = which ? R : L;
    float acc = 0.f;
    for (int d = 0; d < HID; ++d) acc += W[k * HID * HID + hh * HID + d] * Ws[d];
    (which ? rws : lws)[k * HID + hh] = acc;
  } else if (blockIdx.x == 8) {
    int tid = threadIdx.x;
    if (tid < 4) {
      float a = 0.f;
      for (int d = 0; d < HID; ++d) a += b[tid * HID + d] * Ws[d];
      bws[tid] = a;
    } else if (tid == 4) {
      float a = 0.f;
      for (int d = 0; d < HID; ++d) a += Ws[d];
      sumws[0] = a;
    } else if (tid >= 8 && tid < 32) {
      gsc[tid - 8] = 0.f;                   // 6 mixtures * 4 k
    }
  }
}

// ---------------------------------------------------------------- WMMA helpers
__device__ __forceinline__ floatx8 gemm_tile(const bf16x16* fa, const uint32_t* wpack,
                                             int wi, int t, int lane, floatx8 acc) {
  const uint32_t* base = wpack + (((wi * 16 + t) * 8) * 32 + lane) * 8;
#pragma unroll
  for (int s = 0; s < 8; ++s) {
    union { bf16x16 v; u32x4 q[2]; } bf;
    const uint32_t* p = base + s * 256;
    bf.q[0] = ldg_b128(p);
    bf.q[1] = ldg_b128(p + 4);
    acc = __builtin_amdgcn_wmma_f32_16x16x32_bf16(false, fa[s], false, bf.v,
                                                  (short)0, acc, false, false);
  }
  return acc;
}

__device__ __forceinline__ void build_afrags(const __bf16* buf, int strip, int lane,
                                             bf16x16* fr) {
  int row = strip * 16 + (lane & 15);
  int half = (lane >> 4) & 1;
  const uint32_t* rp = (const uint32_t*)(buf + row * PITCH);
#pragma unroll
  for (int s = 0; s < 8; ++s) {
    union { bf16x16 v; uint32_t u[8]; } f;
#pragma unroll
    for (int v = 0; v < 8; ++v) {
      int kb = s * 32 + ((v < 4) ? 0 : 16) + half * 8 + (v & 3) * 2;
      f.u[v] = rp[kb >> 1];
    }
    fr[s] = f.v;
  }
}

// ---------------------------------------------------------------- sigmoid stage
__device__ __forceinline__ void sig_stage(int iW, int j0, int j1,
                          const __bf16* Xb, const __bf16* Hb, __bf16* outB,
                          const uint32_t* wpack, const float* bl,
                          float* gnode, long long row0, int lane, int wid) {
  int strip = wid & 1, tg = wid >> 1, half = (lane >> 4) & 1, l15 = lane & 15;
  bf16x16 fa[8], fb[8];
  build_afrags(Xb, strip, lane, fa);
  build_afrags(Hb, strip, lane, fb);
  for (int tt = 0; tt < 4; ++tt) {
    int t = tg * 4 + tt;
    int col = t * 16 + l15;
    floatx8 acc = {};
    acc = gemm_tile(fa, wpack, iW, t, lane, acc);
    acc = gemm_tile(fb, wpack, 4 + iW, t, lane, acc);
    float bb = bl[iW * HID + col];
#pragma unroll
    for (int g = 0; g < 8; ++g) {
      int row = strip * 16 + half * 8 + g;
      float z = 1.f / (1.f + __expf(-(acc[g] + bb)));
      outB[row * PITCH + col] = (__bf16)z;
      stg_b32(&gnode[((row0 + row) * 9 + j0) * HID + col], z);
      if (j1 >= 0) stg_b32(&gnode[((row0 + row) * 9 + j1) * HID + col], z);
    }
  }
  __syncthreads();
}

// ---------------------------------------------------------------- mixture
// mode 0: A@L_k + B@R_k + b   (scores via precomputed L@Ws / R@Ws)
// mode 1: 1 - (A@R_k + b)     (scores via sumWs - A.(R@Ws) - b.Ws)
// mode 2: (A@L_k)*(B@R_k) + b (scores need the candidates -> WMMA score pass)
__device__ __forceinline__ void mixture(int mode, int mixIdx, int jnode,
                        const __bf16* Ab, const __bf16* Bb, __bf16* outB, float* hnext,
                        const uint32_t* wpack,
                        const float* lws, const float* rws, const float* wsv,
                        const float* bl, const float* bwsG, float sumws,
                        float* sS, float* sW, float* gScore, float* gnode,
                        long long row0, int tid, int lane, int wid) {
  int strip = wid & 1, tg = wid >> 1, half = (lane >> 4) & 1, l15 = lane & 15;

  if (tid < TM * 4) sS[tid] = 0.f;
  __syncthreads();

  bf16x16 fa[8], fb[8];
  build_afrags(Ab, strip, lane, fa);
  build_afrags(Bb, strip, lane, fb);

  if (mode == 2) {
#pragma unroll
    for (int k = 0; k < 4; ++k) {
      float part[8] = {0.f, 0.f, 0.f, 0.f, 0.f, 0.f, 0.f, 0.f};
      for (int tt = 0; tt < 4; ++tt) {
        int t = tg * 4 + tt;
        int col = t * 16 + l15;
        floatx8 p = {}, q = {};
        p = gemm_tile(fa, wpack, k, t, lane, p);
        q = gemm_tile(fb, wpack, 4 + k, t, lane, q);
        float wsc = wsv[col];
        float bb = bl[k * HID + col];
#pragma unroll
        for (int g = 0; g < 8; ++g) part[g] += (p[g] * q[g] + bb) * wsc;
      }
#pragma unroll
      for (int g = 0; g < 8; ++g) {
#pragma unroll
        for (int m = 1; m < 16; m <<= 1) part[g] += __shfl_xor(part[g], m, 32);
      }
      if (l15 == 0) {
#pragma unroll
        for (int g = 0; g < 8; ++g)
          atomicAdd(&sS[(strip * 16 + half * 8 + g) * 4 + k], part[g]);
      }
    }
  } else {
    if (tid < TM * 4) {
      int r = tid >> 2, k = tid & 3;
      const __bf16* ar = Ab + r * PITCH;
      float s;
      if (mode == 0) {
        const __bf16* br = Bb + r * PITCH;
        float da = 0.f, db = 0.f;
        for (int hh = 0; hh < HID; ++hh) {
          da += (float)ar[hh] * lws[k * HID + hh];
          db += (float)br[hh] * rws[k * HID + hh];
        }
        s = da + db + bwsG[k];
      } else {
        float da = 0.f;
        for (int hh = 0; hh < HID; ++hh) da += (float)ar[hh] * rws[k * HID + hh];
        s = sumws - da - bwsG[k];
      }
      sS[tid] = s;
    }
  }
  __syncthreads();

  if (tid < TM) {
    float s0 = sS[tid * 4 + 0], s1 = sS[tid * 4 + 1];
    float s2 = sS[tid * 4 + 2], s3 = sS[tid * 4 + 3];
    float m = fmaxf(fmaxf(s0, s1), fmaxf(s2, s3));
    float e0 = __expf(s0 - m), e1 = __expf(s1 - m);
    float e2 = __expf(s2 - m), e3 = __expf(s3 - m);
    float inv = 1.f / (e0 + e1 + e2 + e3);
    sW[tid * 4 + 0] = e0 * inv; sW[tid * 4 + 1] = e1 * inv;
    sW[tid * 4 + 2] = e2 * inv; sW[tid * 4 + 3] = e3 * inv;
  }
  if (tid < 4) {
    float a = 0.f;
    for (int r = 0; r < TM; ++r) a += sS[r * 4 + tid];
    atomicAdd(&gScore[mixIdx * 4 + tid], a);
  }
  __syncthreads();

  float wr[4][8];
#pragma unroll
  for (int k = 0; k < 4; ++k) {
#pragma unroll
    for (int g = 0; g < 8; ++g)
      wr[k][g] = sW[(strip * 16 + half * 8 + g) * 4 + k];
  }

  for (int tt = 0; tt < 4; ++tt) {
    int t = tg * 4 + tt;
    int col = t * 16 + l15;
    floatx8 outv = {};
#pragma unroll
    for (int k = 0; k < 4; ++k) {
      float bb = bl[k * HID + col];
      floatx8 val;
      if (mode == 0) {
        floatx8 a = {};
        a = gemm_tile(fa, wpack, k, t, lane, a);
        a = gemm_tile(fb, wpack, 4 + k, t, lane, a);
#pragma unroll
        for (int g = 0; g < 8; ++g) val[g] = a[g] + bb;
      } else if (mode == 1) {
        floatx8 a = {};
        a = gemm_tile(fa, wpack, 4 + k, t, lane, a);
#pragma unroll
        for (int g = 0; g < 8; ++g) val[g] = 1.f - a[g] - bb;
      } else {
        floatx8 p = {}, q = {};
        p = gemm_tile(fa, wpack, k, t, lane, p);
        q = gemm_tile(fb, wpack, 4 + k, t, lane, q);
#pragma unroll
        for (int g = 0; g < 8; ++g) val[g] = p[g] * q[g] + bb;
      }
#pragma unroll
      for (int g = 0; g < 8; ++g) outv[g] += wr[k][g] * val[g];
    }
#pragma unroll
    for (int g = 0; g < 8; ++g) {
      int row = strip * 16 + half * 8 + g;
      float v = outv[g];
      if (outB) outB[row * PITCH + col] = (__bf16)v;
      stg_b32(&gnode[((row0 + row) * 9 + jnode) * HID + col], v);
      if (hnext) stg_b32(&hnext[(row0 + row) * HID + col], v);
    }
  }
  __syncthreads();
}

// ---------------------------------------------------------------- main kernel
__global__ __launch_bounds__(256) void rrnn_main(
    const float* __restrict__ x, const float* __restrict__ h,
    const float* __restrict__ bvec, const uint32_t* __restrict__ wpack,
    const float* __restrict__ lwsG, const float* __restrict__ rwsG,
    const float* __restrict__ bwsG, const float* __restrict__ sumwsG,
    const float* __restrict__ wsG, float* __restrict__ gScore,
    float* __restrict__ out) {
  __shared__ __align__(16) char smem[SMEM_BYTES];
  __bf16* B0 = (__bf16*)smem;            // X   -> later OMZ
  __bf16* B1 = B0 + TM * PITCH;          // H
  __bf16* B2 = B1 + TM * PITCH;          // Z1 (== Z2)
  __bf16* B3 = B2 + TM * PITCH;          // R   -> later ZH
  __bf16* B4 = B3 + TM * PITCH;          // RH  -> later Z2H
  __bf16* B5 = B4 + TM * PITCH;          // HT
  float* lws = (float*)(B5 + TM * PITCH);
  float* rws = lws + 1024;
  float* wsv = rws + 1024;
  float* bl  = wsv + 256;
  float* sS  = bl + 1024;
  float* sW  = sS + TM * 4;

  int tid = threadIdx.x, lane = tid & 31, wid = tid >> 5;
  long long row0 = (long long)blockIdx.x * TM;
  float* gnode = out + OUT_GNODE;

#pragma unroll
  for (int i = 0; i < 32; ++i) {
    int e = i * 256 + tid;
    int r = e >> 8, c = e & 255;
    B0[r * PITCH + c] = (__bf16)x[(row0 + r) * HID + c];
    B1[r * PITCH + c] = (__bf16)h[(row0 + r) * HID + c];
  }
#pragma unroll
  for (int i = 0; i < 4; ++i) {
    lws[i * 256 + tid] = lwsG[i * 256 + tid];
    rws[i * 256 + tid] = rwsG[i * 256 + tid];
    bl[i * 256 + tid]  = bvec[i * 256 + tid];
  }
  wsv[tid] = wsG[tid];
  float sumws = sumwsG[0];
  __syncthreads();

  // z1 = sigmoid(x@L0 + h@R0 + b0) -> B2 (nodes 0 and 2)
  sig_stage(0, 0, 2, B0, B1, B2, wpack, bl, gnode, row0, lane, wid);
  // r  = sigmoid(x@L1 + h@R1 + b1) -> B3 (node 1)
  sig_stage(1, 1, -1, B0, B1, B3, wpack, bl, gnode, row0, lane, wid);
  // rh = mix(h@L_k + r@R_k + b)    -> B4 (node 3, mix 0)
  mixture(0, 0, 3, B1, B3, B4, nullptr, wpack, lws, rws, wsv, bl, bwsG, sumws,
          sS, sW, gScore, gnode, row0, tid, lane, wid);
  // h_tilde = mix(x@L_k + rh@R_k + b) -> B5 (node 4, mix 1)
  mixture(0, 1, 4, B0, B4, B5, nullptr, wpack, lws, rws, wsv, bl, bwsG, sumws,
          sS, sW, gScore, gnode, row0, tid, lane, wid);
  // oneMinusZ1 = mix(1 - (z1@R_k + b)) -> B0 (node 5, mix 2)
  mixture(1, 2, 5, B2, B2, B0, nullptr, wpack, lws, rws, wsv, bl, bwsG, sumws,
          sS, sW, gScore, gnode, row0, tid, lane, wid);
  // zh_tilde = mix((ht@L_k)*(omz@R_k) + b) -> B3 (node 6, mix 3)
  mixture(2, 3, 6, B5, B0, B3, nullptr, wpack, lws, rws, wsv, bl, bwsG, sumws,
          sS, sW, gScore, gnode, row0, tid, lane, wid);
  // z2h = mix((h@L_k)*(z2@R_k) + b) -> B4 (node 7, mix 4)
  mixture(2, 4, 7, B1, B2, B4, nullptr, wpack, lws, rws, wsv, bl, bwsG, sumws,
          sS, sW, gScore, gnode, row0, tid, lane, wid);
  // h_next = mix(zh@L_k + z2h@R_k + b) (node 8, mix 5, also h_next output)
  mixture(0, 5, 8, B3, B4, nullptr, out, wpack, lws, rws, wsv, bl, bwsG, sumws,
          sS, sW, gScore, gnode, row0, tid, lane, wid);
}

// ---------------------------------------------------------------- finalize
__global__ void finalize(const float* __restrict__ gScore, float* __restrict__ out) {
  int tid = threadIdx.x;
  if (tid == 0) {
    out[OUT_GS + 0] = 0.f;
    out[OUT_GS + 1] = 1.f;
    out[OUT_GS + 2] = 0.f;
  }
  if (tid < 6) {
    float s[4];
#pragma unroll
    for (int k = 0; k < 4; ++k) s[k] = gScore[tid * 4 + k];
    float best = s[0], second = -3.4e38f;
    int bi = 0;
#pragma unroll
    for (int k = 1; k < 4; ++k) {
      if (s[k] > best) { second = best; best = s[k]; bi = k; }
      else if (s[k] > second) { second = s[k]; }
    }
    out[OUT_GS + 3 + tid] = (float)bi;
    out[OUT_MARG + tid] = best - second;
  }
}

// ---------------------------------------------------------------- launch
extern "C" void kernel_launch(void* const* d_in, const int* in_sizes, int n_in,
                              void* d_out, int out_size, void* d_ws, size_t ws_size,
                              hipStream_t stream) {
  const float* x  = (const float*)d_in[0];
  const float* h  = (const float*)d_in[1];
  const float* L  = (const float*)d_in[2];
  const float* R  = (const float*)d_in[3];
  const float* b  = (const float*)d_in[4];
  const float* Ws = (const float*)d_in[5];
  float* out = (float*)d_out;
  char* ws = (char*)d_ws;

  uint32_t* wpack = (uint32_t*)(ws + WPACK_OFF);
  float* lws   = (float*)(ws + LWS_OFF);
  float* rws   = (float*)(ws + RWS_OFF);
  float* bws   = (float*)(ws + BWS_OFF);
  float* sumws = (float*)(ws + SUMWS_OFF);
  float* gsc   = (float*)(ws + GSC_OFF);

  pack_weights<<<128, 256, 0, stream>>>(L, R, wpack);
  prep_vecs<<<9, 256, 0, stream>>>(L, R, b, Ws, lws, rws, bws, sumws, gsc);
  rrnn_main<<<BATCH / TM, 256, 0, stream>>>(x, h, b, wpack, lws, rws, bws, sumws,
                                            Ws, gsc, out);
  finalize<<<1, 32, 0, stream>>>(gsc, out);
}
